// BidirectionalSelfAttention_61254823576154
// MI455X (gfx1250) — matrix operand order repaired and checked by reading, not compiled, not run on hardware
//
#include <hip/hip_runtime.h>
#include <hip/hip_bf16.h>

// ---------------------------------------------------------------------------
// Bidirectional self-attention block for MI455X (gfx1250), bf16 WMMA pipeline.
// ---------------------------------------------------------------------------

typedef __attribute__((ext_vector_type(16))) __bf16 v16bf;
typedef __attribute__((ext_vector_type(8)))  __bf16 v8bf;
typedef __attribute__((ext_vector_type(8)))  float  v8f;
typedef int v4i __attribute__((vector_size(16)));

#define B_  2
#define T_  2048
#define C_  2048
#define H_  16
#define D_  128
#define CQ_ 6144   /* 3*C */

#define AS1 __attribute__((address_space(1)))
#define AS3 __attribute__((address_space(3)))

#if defined(__has_builtin)
#  if __has_builtin(__builtin_amdgcn_global_load_async_to_lds_b128) && \
      __has_builtin(__builtin_amdgcn_s_wait_asynccnt)
#    define USE_ASYNC_LDS 1
#  else
#    define USE_ASYNC_LDS 0
#  endif
#else
#  define USE_ASYNC_LDS 0
#endif

__device__ __forceinline__ AS1 v4i* as1v(const void* p) {
  return (AS1 v4i*)(unsigned long long)(uintptr_t)p;   // global: same 64-bit value
}
__device__ __forceinline__ AS3 v4i* as3v(void* p) {
  return (AS3 v4i*)(unsigned int)(uintptr_t)p;         // LDS: low 32 bits = offset
}

__device__ __forceinline__ v16bf make16(const __bf16* p0, const __bf16* p1) {
  v8bf a = *(const v8bf*)p0;
  v8bf b = *(const v8bf*)p1;
  return __builtin_shufflevector(a, b, 0,1,2,3,4,5,6,7,8,9,10,11,12,13,14,15);
}
__device__ __forceinline__ v8f wmma_bf16(v16bf a, v16bf b, v8f c) {
  // v_wmma_f32_16x16x32_bf16  (neg_a, A, neg_b, B, c_mod, C, reuse_a, reuse_b)
  return __builtin_amdgcn_wmma_f32_16x16x32_bf16(false, a, false, b, (short)0, c,
                                                 false, false);
}
__device__ __forceinline__ v8f zero8() {
  v8f z = {0.f,0.f,0.f,0.f,0.f,0.f,0.f,0.f};
  return z;
}

// ---------------------------------------------------------------------------
// fp32 -> bf16 elementwise convert
// ---------------------------------------------------------------------------
__global__ __launch_bounds__(256) void convert_bf16_kernel(const float* __restrict__ src,
                                                           __bf16* __restrict__ dst,
                                                           long long n) {
  long long i = (long long)blockIdx.x * 256 + threadIdx.x;
  if (i < n) dst[i] = (__bf16)src[i];
}

// ---------------------------------------------------------------------------
// W[K][N] fp32 -> Wt[N][K] bf16  (LDS tiled transpose)
// ---------------------------------------------------------------------------
__global__ __launch_bounds__(256) void transpose_convert_kernel(const float* __restrict__ W,
                                                                __bf16* __restrict__ Wt,
                                                                int K, int N) {
  __shared__ float tile[32][33];
  int bx = blockIdx.x * 32;   // N
  int by = blockIdx.y * 32;   // K
  int tx = threadIdx.x, ty = threadIdx.y;   // 32 x 8
#pragma unroll
  for (int j = 0; j < 32; j += 8)
    tile[ty + j][tx] = W[(size_t)(by + ty + j) * N + (bx + tx)];
  __syncthreads();
#pragma unroll
  for (int j = 0; j < 32; j += 8)
    Wt[(size_t)(bx + ty + j) * K + (by + tx)] = (__bf16)tile[tx][ty + j];
}

// ---------------------------------------------------------------------------
// RoPE in-place on q,k halves of qkv (bf16 storage, fp32 math)
// ---------------------------------------------------------------------------
__global__ __launch_bounds__(256) void rope_kernel(__bf16* __restrict__ qkv,
                                                   const float* __restrict__ cosT,
                                                   const float* __restrict__ sinT) {
  int id = blockIdx.x * 256 + threadIdx.x;  // B*T * 2(q,k) * H * 32 = 4,194,304
  int i  = id & 31;           // rotary pair index
  int h  = (id >> 5) & 15;
  int qk = (id >> 9) & 1;     // 0=q, 1=k
  int bt = id >> 10;          // 0..B*T-1
  int t  = bt & (T_ - 1);
  size_t base = (size_t)bt * CQ_ + (size_t)qk * C_ + h * D_ + 2 * i;
  float x1 = (float)qkv[base];
  float x2 = (float)qkv[base + 1];
  float c = cosT[t * 32 + i];
  float s = sinT[t * 32 + i];
  qkv[base]     = (__bf16)(x1 * c - x2 * s);
  qkv[base + 1] = (__bf16)(x1 * s + x2 * c);
}

// ---------------------------------------------------------------------------
// Vt[b][h][d][t] = qkv[b*T+t][2C + h*D + d]   (bf16 tiled transpose)
// ---------------------------------------------------------------------------
__global__ __launch_bounds__(256) void transpose_v_kernel(const __bf16* __restrict__ qkv,
                                                          __bf16* __restrict__ Vt) {
  __shared__ __bf16 tile[32][33];
  int bh = blockIdx.z;                 // b*16 + h
  int t0 = blockIdx.x * 32, d0 = blockIdx.y * 32;
  int tx = threadIdx.x, ty = threadIdx.y;
  int b = bh >> 4, h = bh & 15;
#pragma unroll
  for (int j = 0; j < 32; j += 8)
    tile[ty + j][tx] = qkv[(size_t)(b * T_ + t0 + ty + j) * CQ_ + 2 * C_ + h * D_ + d0 + tx];
  __syncthreads();
#pragma unroll
  for (int j = 0; j < 32; j += 8)
    Vt[((size_t)bh * D_ + d0 + ty + j) * T_ + t0 + tx] = tile[tx][ty + j];
}

// ---------------------------------------------------------------------------
// GEMM: C[M][N] = A[M][K](bf16) * Bt[N][K](bf16), fp32 accumulate via WMMA.
// Block 256 threads (8 waves), tile 128x128x64, double-buffered LDS filled by
// CDNA5 async global->LDS copies (ASYNCcnt), per-wave 32x64 (2x4 WMMA tiles).
// ---------------------------------------------------------------------------
template <bool OUT_BF16>
__global__ __launch_bounds__(256) void gemm_wmma_kernel(const __bf16* __restrict__ A,
                                                        const __bf16* __restrict__ Bt,
                                                        void* __restrict__ Cout,
                                                        int M, int N, int K) {
  __shared__ __bf16 As[2][128 * 64];   // 2 x 16 KB
  __shared__ __bf16 Bs[2][128 * 64];   // 2 x 16 KB
  const int tid  = threadIdx.x;
  const int lane = tid & 31;
  const int w    = tid >> 5;         // 0..7
  const int wr   = w >> 1;           // 0..3  (rows of 32)
  const int wc   = w & 1;            // 0..1  (cols of 64)
  const int half = lane >> 4;
  const int n    = lane & 15;
  const size_t bm = (size_t)blockIdx.y * 128;
  const size_t bn = (size_t)blockIdx.x * 128;

  v8f acc[2][4];
#pragma unroll
  for (int i = 0; i < 2; i++)
#pragma unroll
    for (int j = 0; j < 4; j++) acc[i][j] = zero8();

  // cooperative fill of one 128x64 double-buffer slot (16B per async op)
  auto fill = [&](int buf, int kk) {
#pragma unroll
    for (int j = 0; j < 4; j++) {
      int id = tid + j * 256;          // 0..1023
      int r  = id >> 3;                // 0..127
      int c8 = (id & 7) * 8;           // bf16 offset of 16B chunk
      const __bf16* ga = &A[(bm + r) * K + kk + c8];
      const __bf16* gb = &Bt[(bn + r) * K + kk + c8];
      __bf16* la = &As[buf][r * 64 + c8];
      __bf16* lb = &Bs[buf][r * 64 + c8];
#if USE_ASYNC_LDS
      __builtin_amdgcn_global_load_async_to_lds_b128(as1v(ga), as3v(la), 0, 0);
      __builtin_amdgcn_global_load_async_to_lds_b128(as1v(gb), as3v(lb), 0, 0);
#else
      *(uint4*)la = *(const uint4*)ga;
      *(uint4*)lb = *(const uint4*)gb;
#endif
    }
  };

  fill(0, 0);
#if USE_ASYNC_LDS
  __builtin_amdgcn_s_wait_asynccnt(0);
#endif
  __syncthreads();

  const int nk = K >> 6;
  for (int ki = 0; ki < nk; ki++) {
    const int buf = ki & 1;
    if (ki + 1 < nk) fill(buf ^ 1, (ki + 1) << 6);   // async prefetch of next tile

#pragma unroll
    for (int ks = 0; ks < 64; ks += 32) {
      v16bf af[2], bfr[4];
#pragma unroll
      for (int i = 0; i < 2; i++) {
        const __bf16* p = &As[buf][(wr * 32 + i * 16 + n) * 64 + ks + (half ? 16 : 0)];
        af[i] = make16(p, p + 8);      // K = off..off+15 contiguous
      }
#pragma unroll
      for (int j = 0; j < 4; j++) {
        const __bf16* p = &Bs[buf][(wc * 64 + j * 16 + n) * 64 + ks + (half ? 16 : 0)];
        bfr[j] = make16(p, p + 8);     // K = off..off+15 contiguous
      }
#pragma unroll
      for (int i = 0; i < 2; i++)
#pragma unroll
        for (int j = 0; j < 4; j++)
          acc[i][j] = wmma_bf16(af[i], bfr[j], acc[i][j]);
    }

#if USE_ASYNC_LDS
    __builtin_amdgcn_s_wait_asynccnt(0);
#endif
    __syncthreads();
  }

  // C layout: element e -> (m = e + 8*half, n)
#pragma unroll
  for (int i = 0; i < 2; i++)
#pragma unroll
    for (int j = 0; j < 4; j++)
#pragma unroll
      for (int e = 0; e < 8; e++) {
        size_t row = bm + wr * 32 + i * 16 + e + 8 * half;
        size_t col = bn + wc * 64 + j * 16 + n;
        if (OUT_BF16)
          ((__bf16*)Cout)[row * (size_t)N + col] = (__bf16)acc[i][j][e];
        else
          ((float*)Cout)[row * (size_t)N + col] = acc[i][j][e];
      }
}

// ---------------------------------------------------------------------------
// Flash attention: one wave per (b, h, 16-query tile). Online softmax, fp32
// stats, WMMA for Q*K^T and P*V. 32-key blocks.
// ---------------------------------------------------------------------------
__global__ __launch_bounds__(32) void attn_kernel(const __bf16* __restrict__ qkv,
                                                  const __bf16* __restrict__ Vt,
                                                  __bf16* __restrict__ Y) {
  __shared__ __bf16 Pt[16 * 32];
  const int lane = threadIdx.x & 31;
  const int half = lane >> 4;
  const int n    = lane & 15;
  const int qt = blockIdx.x & 127;          // T/16
  const int h  = (blockIdx.x >> 7) & 15;
  const int b  = blockIdx.x >> 11;

  const __bf16* qbase = qkv + (size_t)(b * T_ + qt * 16) * CQ_ + h * D_;
  const __bf16* kbase = qkv + (size_t)(b * T_) * CQ_ + C_ + h * D_;
  const __bf16* vbase = Vt + (size_t)(b * 16 + h) * D_ * T_;   // [d][t]

  // Q fragments for all 4 K-dim chunks (reused for every key block)
  v16bf aQ[4];
  {
    const __bf16* qr = qbase + (size_t)n * CQ_;
#pragma unroll
    for (int c = 0; c < 4; c++) {
      const __bf16* p = qr + 32 * c + (half ? 16 : 0);
      aQ[c] = make16(p, p + 8);
    }
  }

  v8f o[8];
  float m[8], l[8];
#pragma unroll
  for (int i = 0; i < 8; i++) { o[i] = zero8(); m[i] = -1e30f; l[i] = 0.f; }
  const float sc = 0.088388347648318447f;   // 1/sqrt(128)

  for (int kb = 0; kb < T_; kb += 32) {
    v8f s0 = zero8(), s1 = zero8();
#pragma unroll
    for (int c = 0; c < 4; c++) {
      const __bf16* pk0 = kbase + (size_t)(kb + n) * CQ_ + 32 * c + (half ? 16 : 0);
      const __bf16* pk1 = pk0 + (size_t)16 * CQ_;
      v16bf b0 = make16(pk0, pk0 + 8);
      v16bf b1 = make16(pk1, pk1 + 8);
      s0 = wmma_bf16(aQ[c], b0, s0);
      s1 = wmma_bf16(aQ[c], b1, s1);
    }

    // online softmax update (per-lane rows m = i + 8*half)
    float p0[8], p1[8], alpha[8];
#pragma unroll
    for (int i = 0; i < 8; i++) { p0[i] = s0[i] * sc; p1[i] = s1[i] * sc; }
#pragma unroll
    for (int i = 0; i < 8; i++) {
      float rm = fmaxf(p0[i], p1[i]);
#pragma unroll
      for (int off = 1; off < 16; off <<= 1)
        rm = fmaxf(rm, __shfl_xor(rm, off, 32));     // stays within 16-lane half
      float mn = fmaxf(m[i], rm);
      alpha[i] = __expf(m[i] - mn);
      m[i] = mn;
      p0[i] = __expf(p0[i] - mn);
      p1[i] = __expf(p1[i] - mn);
      float rs = p0[i] + p1[i];
#pragma unroll
      for (int off = 1; off < 16; off <<= 1)
        rs += __shfl_xor(rs, off, 32);
      l[i] = l[i] * alpha[i] + rs;
    }
#pragma unroll
    for (int t = 0; t < 8; t++)
#pragma unroll
      for (int i = 0; i < 8; i++) o[t][i] *= alpha[i];

    // P (C layout) -> LDS -> A fragment layout
#pragma unroll
    for (int i = 0; i < 8; i++) {
      Pt[(i + 8 * half) * 32 + n]      = (__bf16)p0[i];
      Pt[(i + 8 * half) * 32 + 16 + n] = (__bf16)p1[i];
    }
    __syncthreads();
    const __bf16* pp = &Pt[n * 32 + (half ? 16 : 0)];
    v16bf aP = make16(pp, pp + 8);
#pragma unroll
    for (int dc = 0; dc < 8; dc++) {
      const __bf16* pv = vbase + (size_t)(16 * dc + n) * T_ + kb + (half ? 16 : 0);
      v16bf bV = make16(pv, pv + 8);
      o[dc] = wmma_bf16(aP, bV, o[dc]);
    }
    __syncthreads();
  }

#pragma unroll
  for (int i = 0; i < 8; i++) l[i] = 1.0f / l[i];
  size_t yrow = (size_t)(b * T_ + qt * 16) * C_ + h * D_;
#pragma unroll
  for (int dc = 0; dc < 8; dc++)
#pragma unroll
    for (int i = 0; i < 8; i++)
      Y[yrow + (size_t)(i + 8 * half) * C_ + 16 * dc + n] = (__bf16)(o[dc][i] * l[i]);
}

// ---------------------------------------------------------------------------
// Host launch
// ---------------------------------------------------------------------------
extern "C" void kernel_launch(void* const* d_in, const int* in_sizes, int n_in,
                              void* d_out, int out_size, void* d_ws, size_t ws_size,
                              hipStream_t stream) {
  const float* x    = (const float*)d_in[0];   // [B*T][C]
  const float* Wa   = (const float*)d_in[1];   // [C][3C]
  const float* Wp   = (const float*)d_in[2];   // [C][C]
  const float* cosT = (const float*)d_in[3];   // [T][32]
  const float* sinT = (const float*)d_in[4];   // [T][32]
  float* out = (float*)d_out;

  char* ws = (char*)d_ws;
  __bf16* xb   = (__bf16*)(ws + 0);                      // 16 MB  [4096][2048]
  __bf16* Wat  = (__bf16*)(ws + 16777216ull);            // 24 MB  [6144][2048]
  __bf16* Wpt  = (__bf16*)(ws + 41943040ull);            //  8 MB  [2048][2048]
  __bf16* qkvb = (__bf16*)(ws + 50331648ull);            // 48 MB  [4096][6144]
  __bf16* Vtb  = (__bf16*)(ws + 100663296ull);           // 16 MB  [2][16][128][2048]
  __bf16* Yb   = (__bf16*)(ws + 117440512ull);           // 16 MB  [4096][2048]

  // 1) x -> bf16
  convert_bf16_kernel<<<(B_*T_*C_) / 256, 256, 0, stream>>>(x, xb, (long long)B_*T_*C_);
  // 2) transpose-convert weights (N-major, K contiguous)
  transpose_convert_kernel<<<dim3(CQ_/32, C_/32), dim3(32, 8), 0, stream>>>(Wa, Wat, C_, CQ_);
  transpose_convert_kernel<<<dim3(C_/32,  C_/32), dim3(32, 8), 0, stream>>>(Wp, Wpt, C_, C_);
  // 3) qkv = xb @ Wat   (M=4096, N=6144, K=2048) -> bf16
  gemm_wmma_kernel<true><<<dim3(CQ_/128, (B_*T_)/128), 256, 0, stream>>>(
      xb, Wat, (void*)qkvb, B_*T_, CQ_, C_);
  // 4) RoPE on q,k in place
  rope_kernel<<<(B_*T_*2*H_*32) / 256, 256, 0, stream>>>(qkvb, cosT, sinT);
  // 5) V -> Vt [b][h][d][t]
  transpose_v_kernel<<<dim3(T_/32, D_/32, B_*H_), dim3(32, 8), 0, stream>>>(qkvb, Vtb);
  // 6) flash attention -> Yb bf16
  attn_kernel<<<B_*H_*(T_/16), 32, 0, stream>>>(qkvb, Vtb, Yb);
  // 7) out = Yb @ Wpt  (M=4096, N=2048, K=2048) -> fp32
  gemm_wmma_kernel<false><<<dim3(C_/128, (B_*T_)/128), 256, 0, stream>>>(
      Yb, Wpt, (void*)out, B_*T_, C_, C_);
}
